// MultiHeadDotProductAttention_9603546874241
// MI455X (gfx1250) — matrix operand order repaired and checked by reading, not compiled, not run on hardware
//
#include <hip/hip_runtime.h>

// ---------------------------------------------------------------------------
// MI455X (gfx1250, wave32) multi-head attention.
// All GEMMs use v_wmma_f32_16x16x32_bf16 (f32 accumulation).
// Bias/K/V streams are prefetched with global_prefetch_b8.
// ---------------------------------------------------------------------------

typedef __bf16        v16bf __attribute__((ext_vector_type(16)));
typedef float         v8f   __attribute__((ext_vector_type(8)));
typedef unsigned int  u32x4 __attribute__((ext_vector_type(4)));

#define B_   4
#define L_   2048
#define H_   16
#define DH_  64
#define DM_  1024
#define ROWS_ (B_ * L_)   // 8192

__device__ __forceinline__ unsigned short f2bf(float f) {
    unsigned int u = __float_as_uint(f);
    unsigned int r = u + 0x7FFFu + ((u >> 16) & 1u);   // round-to-nearest-even
    return (unsigned short)(r >> 16);
}

__device__ __forceinline__ v8f v8f_zero() {
    v8f z;
#pragma unroll
    for (int i = 0; i < 8; ++i) z[i] = 0.0f;
    return z;
}

__device__ __forceinline__ v8f wmma_bf16(v16bf a, v16bf b, v8f c) {
    // (neg_a, A, neg_b, B, c_mod, C, reuse_a, reuse_b)
    return __builtin_amdgcn_wmma_f32_16x16x32_bf16(false, a, false, b, (short)0, c,
                                                   false, false);
}

// A-fragment (16x32 bf16, lane holds row M=lane&15):
//   elements 0..7  -> K = kbase + (lane>>4)*8 + 0..7
//   elements 8..15 -> K = kbase + 16 + (lane>>4)*8 + 0..7
__device__ __forceinline__ v16bf load_a(const unsigned short* p, int stride,
                                        int row0, int kbase) {
    const int lane = threadIdx.x & 31;
    const int m = row0 + (lane & 15);
    const int k = kbase + ((lane >> 4) << 3);
    const unsigned short* q = p + (size_t)m * stride + k;
    union { v16bf v; u32x4 q[2]; } u;
    u.q[0] = *(const u32x4*)(q);
    u.q[1] = *(const u32x4*)(q + 16);
    return u.v;
}

// B-fragment (32x16 bf16, lane holds col N=lane&15):
//   elements 0..7  -> K = kbase + (lane>>4)*8 + 0..7, elements 8..15 -> K = kbase + 16 + (lane>>4)*8 + 0..7 (source stored [N][K], K contig)
__device__ __forceinline__ v16bf load_b(const unsigned short* p, int stride,
                                        int n0, int kbase) {
    const int lane = threadIdx.x & 31;
    const int n = n0 + (lane & 15);
    const int k = kbase + ((lane >> 4) << 3);
    const unsigned short* q = p + (size_t)n * stride + k;
    union { v16bf v; u32x4 q[2]; } u;
    u.q[0] = *(const u32x4*)(q);
    u.q[1] = *(const u32x4*)(q + 16);
    return u.v;
}

// ------------------------------- prep kernels -------------------------------

__global__ void cvt_bf16_kernel(const float* __restrict__ in,
                                unsigned short* __restrict__ out, int n) {
    int i = blockIdx.x * blockDim.x + threadIdx.x;
    if (i < n) out[i] = f2bf(in[i]);
}

// in: [1024][1024] f32 row-major (K rows, out-col cols) -> out[c][k] bf16
__global__ void transpose_bf16_kernel(const float* __restrict__ in,
                                      unsigned short* __restrict__ out) {
    int i = blockIdx.x * blockDim.x + threadIdx.x;   // 1M elements
    int k = i >> 10, c = i & 1023;
    out[(size_t)c * 1024 + k] = f2bf(in[(size_t)k * 1024 + c]);
}

// ------------------------------ QKV projection ------------------------------
// blockIdx.z: 0=Q (scaled 1/8), 1=K, 2=V (stored transposed [dh][L]).
// One wave computes a 16x64 output tile.
__global__ __launch_bounds__(128) void qkv_proj_kernel(
    const unsigned short* __restrict__ Xq, const unsigned short* __restrict__ Xkv,
    const unsigned short* __restrict__ Wtq, const unsigned short* __restrict__ Wtk,
    const unsigned short* __restrict__ Wtv,
    const float* __restrict__ bq, const float* __restrict__ bk,
    const float* __restrict__ bv,
    unsigned short* __restrict__ Qbf, unsigned short* __restrict__ Kbf,
    unsigned short* __restrict__ Vt) {
    const int warp = threadIdx.x >> 5, lane = threadIdx.x & 31;
    const int hi = lane >> 4, ln = lane & 15;
    const int wid = blockIdx.x * 4 + warp;           // 0..8191
    const int rt = wid & 511, cg = wid >> 9;
    const int m = blockIdx.z;
    const unsigned short* X = (m == 0) ? Xq : Xkv;
    const unsigned short* W = (m == 0) ? Wtq : (m == 1) ? Wtk : Wtv;
    const float* bias = (m == 0) ? bq : (m == 1) ? bk : bv;
    const int r0 = rt * 16, c0 = cg * 64;

    v8f acc[4];
#pragma unroll
    for (int ct = 0; ct < 4; ++ct) acc[ct] = v8f_zero();

    for (int kk = 0; kk < DM_; kk += 32) {
        v16bf a = load_a(X, DM_, r0, kk);
#pragma unroll
        for (int ct = 0; ct < 4; ++ct) {
            v16bf bfr = load_b(W, DM_, c0 + ct * 16, kk);
            acc[ct] = wmma_bf16(a, bfr, acc[ct]);
        }
    }

    const int bb = r0 >> 11;           // batch
    const int l0 = r0 & 2047;          // position within batch
#pragma unroll
    for (int ct = 0; ct < 4; ++ct) {
        const int col = c0 + ct * 16 + ln;
        const float bsc = bias[col];
        const int h = col >> 6, d = col & 63;
        if (m == 2) {
            unsigned int w[4];
#pragma unroll
            for (int i = 0; i < 4; ++i) {
                float f0 = acc[ct][2 * i] + bsc;
                float f1 = acc[ct][2 * i + 1] + bsc;
                w[i] = (unsigned)f2bf(f0) | ((unsigned)f2bf(f1) << 16);
            }
            u32x4 pack; pack[0] = w[0]; pack[1] = w[1]; pack[2] = w[2]; pack[3] = w[3];
            *(u32x4*)&Vt[(size_t)((bb * H_ + h) * DH_ + d) * L_ + l0 + 8 * hi] = pack;
        } else {
            unsigned short* OUT = (m == 0) ? Qbf : Kbf;
            const float scale = (m == 0) ? 0.125f : 1.0f;   // 1/sqrt(64)
#pragma unroll
            for (int r = 0; r < 8; ++r) {
                float f = (acc[ct][r] + bsc) * scale;
                OUT[(size_t)((bb * H_ + h) * L_ + l0 + r + 8 * hi) * DH_ + d] = f2bf(f);
            }
        }
    }
}

// ---------------------- flash attention (bias + softmax) --------------------
// One wave handles 32 q-rows of one (b,h); online softmax over 64 k-chunks of 32.
__global__ __launch_bounds__(128) void attn_kernel(
    const unsigned short* __restrict__ Qbf, const unsigned short* __restrict__ Kbf,
    const unsigned short* __restrict__ Vt, const float* __restrict__ bias,
    unsigned short* __restrict__ ctx) {
    const int warp = threadIdx.x >> 5, lane = threadIdx.x & 31;
    const int hi = lane >> 4, ln = lane & 15;
    const int wid = blockIdx.x * 4 + warp;           // 0..4095
    const int qt = wid & 63;
    const int h = (wid >> 6) & 15;
    const int b = wid >> 10;

    const unsigned short* Qh = Qbf + (size_t)(b * H_ + h) * L_ * DH_;
    const unsigned short* Kh = Kbf + (size_t)(b * H_ + h) * L_ * DH_;
    const unsigned short* Vh = Vt + (size_t)(b * H_ + h) * DH_ * L_;
    const float* biash = bias + (size_t)h * L_ * L_;
    const int q0 = qt * 32;

    __shared__ unsigned short plds[4][32 * 32];      // per-wave P staging (bf16)
    unsigned short* myp = plds[warp];

    v16bf qf[2][2];
#pragma unroll
    for (int rt = 0; rt < 2; ++rt)
#pragma unroll
        for (int ks = 0; ks < 2; ++ks)
            qf[rt][ks] = load_a(Qh, DH_, q0 + rt * 16, ks * 32);

    v8f o[2][4];
    float mrow[2][8], lrow[2][8];
#pragma unroll
    for (int rt = 0; rt < 2; ++rt) {
#pragma unroll
        for (int dt = 0; dt < 4; ++dt) o[rt][dt] = v8f_zero();
#pragma unroll
        for (int r = 0; r < 8; ++r) { mrow[rt][r] = -1e30f; lrow[rt][r] = 0.0f; }
    }

    for (int kc = 0; kc < 64; ++kc) {
        const int k0 = kc * 32;

        // Prefetch the HBM-resident streams 4 chunks ahead:
        // bias: 32 rows x 2 cachelines; K: 32 rows x 2 lines; V: 64 dh-rows x 1 line.
        const int k0n = k0 + 128;
        if (k0n < L_) {
            const float* bp = &biash[(size_t)(q0 + lane) * L_ + k0n];
            __builtin_prefetch(bp, 0, 1);
            __builtin_prefetch(bp + 16, 0, 1);
            const unsigned short* kp = &Kh[(size_t)(k0n + lane) * DH_];
            __builtin_prefetch(kp, 0, 1);
            __builtin_prefetch(kp + 32, 0, 1);
            __builtin_prefetch(&Vh[(size_t)lane * L_ + k0n], 0, 1);
            __builtin_prefetch(&Vh[(size_t)(lane + 32) * L_ + k0n], 0, 1);
        }

        // S = Q * K^T  (two 16-col subtiles)
        v8f s[2][2];
#pragma unroll
        for (int ct = 0; ct < 2; ++ct) {
            v16bf kf0 = load_b(Kh, DH_, k0 + ct * 16, 0);
            v16bf kf1 = load_b(Kh, DH_, k0 + ct * 16, 32);
#pragma unroll
            for (int rt = 0; rt < 2; ++rt) {
                v8f a = v8f_zero();
                a = wmma_bf16(qf[rt][0], kf0, a);
                a = wmma_bf16(qf[rt][1], kf1, a);
                s[rt][ct] = a;
            }
        }
        // bias add (C layout: lane=col, element r = row r + 8*hi)
#pragma unroll
        for (int rt = 0; rt < 2; ++rt)
#pragma unroll
            for (int ct = 0; ct < 2; ++ct)
#pragma unroll
                for (int r = 0; r < 8; ++r)
                    s[rt][ct][r] +=
                        biash[(size_t)(q0 + rt * 16 + r + 8 * hi) * L_ +
                              (k0 + ct * 16 + ln)];
        // online softmax; rows live across 16 lanes
#pragma unroll
        for (int rt = 0; rt < 2; ++rt) {
#pragma unroll
            for (int r = 0; r < 8; ++r) {
                float mx = fmaxf(s[rt][0][r], s[rt][1][r]);
#pragma unroll
                for (int msk = 1; msk < 16; msk <<= 1)
                    mx = fmaxf(mx, __shfl_xor(mx, msk, 32));
                float mold = mrow[rt][r];
                float mnew = fmaxf(mold, mx);
                float corr = __expf(mold - mnew);
                float p0 = __expf(s[rt][0][r] - mnew);
                float p1 = __expf(s[rt][1][r] - mnew);
                s[rt][0][r] = p0; s[rt][1][r] = p1;
                float rs = p0 + p1;
#pragma unroll
                for (int msk = 1; msk < 16; msk <<= 1)
                    rs += __shfl_xor(rs, msk, 32);
                lrow[rt][r] = lrow[rt][r] * corr + rs;
                mrow[rt][r] = mnew;
#pragma unroll
                for (int dt = 0; dt < 4; ++dt) o[rt][dt][r] *= corr;
            }
        }
        // C-layout -> A-layout through LDS (bf16)
#pragma unroll
        for (int rt = 0; rt < 2; ++rt)
#pragma unroll
            for (int ct = 0; ct < 2; ++ct)
#pragma unroll
                for (int r = 0; r < 8; ++r)
                    myp[(rt * 16 + r + 8 * hi) * 32 + ct * 16 + ln] =
                        f2bf(s[rt][ct][r]);
        __syncthreads();
        v16bf pf[2];
#pragma unroll
        for (int rt = 0; rt < 2; ++rt) {
            const int row = rt * 16 + ln;
            const int kb = hi * 8;
            union { v16bf v; u32x4 q[2]; } u;
            u.q[0] = *(const u32x4*)&myp[row * 32 + kb];
            u.q[1] = *(const u32x4*)&myp[row * 32 + kb + 16];
            pf[rt] = u.v;
        }
        // O += P * V   (V stored transposed: [dh][L], K contiguous)
#pragma unroll
        for (int dt = 0; dt < 4; ++dt) {
            v16bf vf = load_b(Vh, L_, dt * 16, k0);
#pragma unroll
            for (int rt = 0; rt < 2; ++rt)
                o[rt][dt] = wmma_bf16(pf[rt], vf, o[rt][dt]);
        }
        __syncthreads();
    }
    // normalize and store ctx[b,l,h*64+d] as bf16
#pragma unroll
    for (int rt = 0; rt < 2; ++rt)
#pragma unroll
        for (int r = 0; r < 8; ++r) {
            const float inv = 1.0f / lrow[rt][r];
            const size_t row = (size_t)(b * L_ + q0 + rt * 16 + r + 8 * hi);
#pragma unroll
            for (int dt = 0; dt < 4; ++dt)
                ctx[row * DM_ + h * DH_ + dt * 16 + ln] = f2bf(o[rt][dt][r] * inv);
        }
}

// ------------------------------ output projection ---------------------------

__global__ __launch_bounds__(128) void out_proj_kernel(
    const unsigned short* __restrict__ CTX, const unsigned short* __restrict__ Wto,
    const float* __restrict__ bo, float* __restrict__ out) {
    const int warp = threadIdx.x >> 5, lane = threadIdx.x & 31;
    const int hi = lane >> 4, ln = lane & 15;
    const int wid = blockIdx.x * 4 + warp;           // 0..8191
    const int rt = wid & 511, cg = wid >> 9;
    const int r0 = rt * 16, c0 = cg * 64;

    v8f acc[4];
#pragma unroll
    for (int ct = 0; ct < 4; ++ct) acc[ct] = v8f_zero();

    for (int kk = 0; kk < DM_; kk += 32) {
        v16bf a = load_a(CTX, DM_, r0, kk);
#pragma unroll
        for (int ct = 0; ct < 4; ++ct) {
            v16bf bfr = load_b(Wto, DM_, c0 + ct * 16, kk);
            acc[ct] = wmma_bf16(a, bfr, acc[ct]);
        }
    }
#pragma unroll
    for (int ct = 0; ct < 4; ++ct) {
        const int col = c0 + ct * 16 + ln;
        const float bb = bo[col];
#pragma unroll
        for (int r = 0; r < 8; ++r)
            out[(size_t)(r0 + r + 8 * hi) * DM_ + col] = acc[ct][r] + bb;
    }
}

// --------------------------------- launcher ---------------------------------

extern "C" void kernel_launch(void* const* d_in, const int* in_sizes, int n_in,
                              void* d_out, int out_size, void* d_ws, size_t ws_size,
                              hipStream_t stream) {
    const float* inputs_q  = (const float*)d_in[0];
    const float* inputs_kv = (const float*)d_in[1];
    const float* bias      = (const float*)d_in[2];
    const float* wq        = (const float*)d_in[3];
    const float* bq        = (const float*)d_in[4];
    const float* wk        = (const float*)d_in[5];
    const float* bk        = (const float*)d_in[6];
    const float* wv        = (const float*)d_in[7];
    const float* bv        = (const float*)d_in[8];
    const float* wo        = (const float*)d_in[9];
    const float* bo        = (const float*)d_in[10];
    float* out = (float*)d_out;

    char* ws = (char*)d_ws;
    const size_t ACT = (size_t)ROWS_ * DM_ * 2;      // 16 MB bf16 activations
    const size_t WSZ = (size_t)DM_ * DM_ * 2;        //  2 MB bf16 weights
    unsigned short* XQ  = (unsigned short*)(ws);
    unsigned short* XKV = (unsigned short*)(ws + ACT);
    unsigned short* WTQ = (unsigned short*)(ws + 2 * ACT);
    unsigned short* WTK = (unsigned short*)(ws + 2 * ACT + WSZ);
    unsigned short* WTV = (unsigned short*)(ws + 2 * ACT + 2 * WSZ);
    unsigned short* WTO = (unsigned short*)(ws + 2 * ACT + 3 * WSZ);
    unsigned short* QBF = (unsigned short*)(ws + 2 * ACT + 4 * WSZ);
    unsigned short* KBF = (unsigned short*)(ws + 3 * ACT + 4 * WSZ);
    unsigned short* VT  = (unsigned short*)(ws + 4 * ACT + 4 * WSZ);
    unsigned short* CTX = (unsigned short*)(ws + 5 * ACT + 4 * WSZ);
    // total: 6*16MB + 4*2MB = 104 MB

    const int nact = ROWS_ * DM_;                    // 8,388,608
    cvt_bf16_kernel<<<nact / 256, 256, 0, stream>>>(inputs_q, XQ, nact);
    cvt_bf16_kernel<<<nact / 256, 256, 0, stream>>>(inputs_kv, XKV, nact);
    transpose_bf16_kernel<<<(DM_ * DM_) / 256, 256, 0, stream>>>(wq, WTQ);
    transpose_bf16_kernel<<<(DM_ * DM_) / 256, 256, 0, stream>>>(wk, WTK);
    transpose_bf16_kernel<<<(DM_ * DM_) / 256, 256, 0, stream>>>(wv, WTV);
    transpose_bf16_kernel<<<(DM_ * DM_) / 256, 256, 0, stream>>>(wo, WTO);

    qkv_proj_kernel<<<dim3(2048, 1, 3), 128, 0, stream>>>(
        XQ, XKV, WTQ, WTK, WTV, bq, bk, bv, QBF, KBF, VT);

    attn_kernel<<<1024, 128, 0, stream>>>(QBF, KBF, VT, bias, CTX);

    out_proj_kernel<<<2048, 128, 0, stream>>>(CTX, WTO, bo, out);
}
